// DotProductAttention_68934225101428
// MI455X (gfx1250) — matrix operand order repaired and checked by reading, not compiled, not run on hardware
//
#include <hip/hip_runtime.h>

typedef __attribute__((ext_vector_type(16))) _Float16 v16h;
typedef __attribute__((ext_vector_type(8)))  _Float16 v8h;
typedef __attribute__((ext_vector_type(8)))  float    v8f;
typedef __attribute__((ext_vector_type(4)))  float    v4f;

#define BATCH 8
#define SEQ   2048
#define DIM   512

// ---------------------------------------------------------------------------
// Kernel 1: projection  Y = f16( X @ W^T ),  X:[16384,512] f32, W:[512,512] f32
// One wave per 32x64 macro-tile = 2x4 WMMA tiles: A frags reused 4x, B frags
// reused 2x -> 8 wmma per 6 fragment loads. 16 K-steps of 32.
// blockIdx.y: 0 -> (queries, Wq, qh), 1 -> (keys, Wk, kh)
// ---------------------------------------------------------------------------
__global__ __launch_bounds__(256) void dpa_proj_kernel(
    const float* __restrict__ Q, const float* __restrict__ K,
    const float* __restrict__ Wq, const float* __restrict__ Wk,
    _Float16* __restrict__ Yq, _Float16* __restrict__ Yk)
{
  const float* X = blockIdx.y ? K  : Q;
  const float* W = blockIdx.y ? Wk : Wq;
  _Float16*    Y = blockIdx.y ? Yk : Yq;

  int wave = (blockIdx.x * 256 + threadIdx.x) >> 5;
  int lane = threadIdx.x & 31;
  int l16  = lane & 15;
  int half = lane >> 4;
  int nt4  = wave & 7;         // 8 macro-tiles over N=512 (64 wide)
  int mt2  = wave >> 3;        // 512 macro-tiles over M=16384 (32 high)

  v8f acc[2][4] = {};

  for (int kk = 0; kk < DIM; kk += 32) {
    v16h a[2], bf[4];
#pragma unroll
    for (int mi = 0; mi < 2; ++mi) {
      int arow = mt2 * 32 + mi * 16 + l16;
      // A fragment: K = kb..kb+7 and kb+16..kb+23 (kb = kk + half*8)
      const v4f* ap = (const v4f*)(X + (size_t)arow * DIM + kk + half * 8);
      v4f a0 = ap[0], a1 = ap[1], a2 = ap[4], a3 = ap[5];
#pragma unroll
      for (int i = 0; i < 4; ++i) {
        a[mi][i]      = (_Float16)a0[i];
        a[mi][i + 4]  = (_Float16)a1[i];
        a[mi][i + 8]  = (_Float16)a2[i];
        a[mi][i + 12] = (_Float16)a3[i];
      }
    }
#pragma unroll
    for (int ni = 0; ni < 4; ++ni) {
      int brow = nt4 * 64 + ni * 16 + l16;   // W row = output column
      // B fragment: K = kb..kb+7 and kb+16..kb+23 (kb = kk + half*8)
      const v4f* bp = (const v4f*)(W + (size_t)brow * DIM + kk + half * 8);
      v4f b0 = bp[0], b1 = bp[1], b2 = bp[4], b3 = bp[5];
#pragma unroll
      for (int i = 0; i < 4; ++i) {
        bf[ni][i]      = (_Float16)b0[i];
        bf[ni][i + 4]  = (_Float16)b1[i];
        bf[ni][i + 8]  = (_Float16)b2[i];
        bf[ni][i + 12] = (_Float16)b3[i];
      }
    }
#pragma unroll
    for (int mi = 0; mi < 2; ++mi)
#pragma unroll
      for (int ni = 0; ni < 4; ++ni)
        acc[mi][ni] = __builtin_amdgcn_wmma_f32_16x16x32_f16(
            false, a[mi], false, bf[ni], (short)0, acc[mi][ni], false, false);
  }

#pragma unroll
  for (int mi = 0; mi < 2; ++mi)
#pragma unroll
    for (int ni = 0; ni < 4; ++ni) {
      int ncol = nt4 * 64 + ni * 16 + l16;
#pragma unroll
      for (int r = 0; r < 8; ++r) {
        int m = mt2 * 32 + mi * 16 + r + half * 8;
        Y[(size_t)m * DIM + ncol] = (_Float16)acc[mi][ni][r];
      }
    }
}

// ---------------------------------------------------------------------------
// Kernel 2: V [B,L,D] f32  ->  VT [B,D,L] f16  (LDS-tiled transpose)
// ---------------------------------------------------------------------------
__global__ __launch_bounds__(256) void dpa_vconv_kernel(
    const float* __restrict__ V, _Float16* __restrict__ VT)
{
  __shared__ _Float16 tile[32][33];
  int b  = blockIdx.z;
  int l0 = blockIdx.x * 32;
  int d0 = blockIdx.y * 32;
  int tx = threadIdx.x & 31;
  int ty = threadIdx.x >> 5;                 // 0..7
  const float* src = V + ((size_t)b * SEQ + l0) * DIM + d0;
#pragma unroll
  for (int i = 0; i < 4; ++i) {
    int l = ty + i * 8;
    tile[l][tx] = (_Float16)src[(size_t)l * DIM + tx];
  }
  __syncthreads();
  _Float16* dst = VT + ((size_t)b * DIM + d0) * SEQ + l0;
#pragma unroll
  for (int i = 0; i < 4; ++i) {
    int d = ty + i * 8;
    dst[(size_t)d * SEQ + tx] = tile[tx][d];
  }
}

// ---------------------------------------------------------------------------
// Kernel 3: flash attention. One block = (batch b, 16 query rows), 8 waves.
// Per 128-key chunk: S = q k^T (WMMA) -> LDS, parallel online softmax
// (256 threads, 2-stage LDS reduction) -> P (f16) in LDS, O += P V (WMMA)
// with alpha-rescaled accumulators. Final O /= l. Prefetch next chunk.
// ---------------------------------------------------------------------------
__global__ __launch_bounds__(256) void dpa_flash_kernel(
    const _Float16* __restrict__ QH, const _Float16* __restrict__ KH,
    const _Float16* __restrict__ VT, float* __restrict__ O)
{
  const int CH = 128;                         // keys per chunk
  __shared__ __align__(16) float    Sbuf[16][CH];
  __shared__ __align__(16) _Float16 Pbuf[16][CH];
  __shared__ float red[16][17];               // partial max / partial sum
  __shared__ float mrow[16], lrow[16], arow[16];

  int b    = blockIdx.y;
  int q0   = blockIdx.x * 16;
  int w    = threadIdx.x >> 5;                // wave 0..7
  int lane = threadIdx.x & 31;
  int l16  = lane & 15;
  int half = lane >> 4;
  int srow = threadIdx.x & 15;                // softmax: row owned by thread
  int sseg = threadIdx.x >> 4;                // softmax: 8-key segment 0..15

  const _Float16* qbase = QH + ((size_t)b * SEQ + q0) * DIM;
  const _Float16* kbase = KH + (size_t)b * SEQ * DIM;
  const _Float16* vbase = VT + (size_t)b * DIM * SEQ;

  // Q fragments for this query tile live in registers (16 K-steps x v16h)
  v16h qf[16];
#pragma unroll
  for (int ks = 0; ks < 16; ++ks) {
    const v8h* qp = (const v8h*)(qbase + (size_t)l16 * DIM + ks * 32 + half * 16);
    v8h lo = qp[0];
    v8h hi = qp[1];                           // +8 halves
#pragma unroll
    for (int i = 0; i < 8; ++i) { qf[ks][i] = lo[i]; qf[ks][i + 8] = hi[i]; }
  }

  if (threadIdx.x < 16) { mrow[threadIdx.x] = -__builtin_inff(); lrow[threadIdx.x] = 0.f; }
  __syncthreads();

  v8f o[4] = {{}, {}, {}, {}};                // wave's 16 x 64 output slice
  int d0 = w * 64;

  for (int c0 = 0; c0 < SEQ; c0 += CH) {
    int krow = c0 + w * 16 + l16;             // key row = B-fragment column

    // prefetch next chunk's K rows and V slice (global_prefetch_b8)
    if (c0 + CH < SEQ) {
      __builtin_prefetch(kbase + (size_t)(krow + CH) * DIM, 0, 1);
      __builtin_prefetch(vbase + (size_t)(d0 + l16) * SEQ + c0 + CH, 0, 1);
    }

    // ---- scores: wave w handles keys [c0 + w*16, +16) ----
    v8f s = {};
#pragma unroll
    for (int ks = 0; ks < 16; ++ks) {
      const v16h* kp = (const v16h*)(kbase + (size_t)krow * DIM + ks * 32 + half * 16);
      v16h bf = kp[0];
      s = __builtin_amdgcn_wmma_f32_16x16x32_f16(false, qf[ks], false, bf,
                                                 (short)0, s, false, false);
    }
#pragma unroll
    for (int r = 0; r < 8; ++r)
      Sbuf[r + half * 8][w * 16 + l16] = s[r];
    __syncthreads();

    // ---- online softmax, parallel over 256 threads ----
    // stage 1: per-(row, 8-key segment) partial max
    {
      float pm = -__builtin_inff();
#pragma unroll
      for (int j = 0; j < 8; ++j) pm = fmaxf(pm, Sbuf[srow][sseg * 8 + j]);
      red[srow][sseg] = pm;
    }
    __syncthreads();
    // stage 2: finalize row max + alpha (16 threads)
    if (threadIdx.x < 16) {
      int row = threadIdx.x;
      float mm = mrow[row];
#pragma unroll
      for (int j = 0; j < 16; ++j) mm = fmaxf(mm, red[row][j]);
      arow[row] = __expf(mrow[row] - mm);
      mrow[row] = mm;
    }
    __syncthreads();
    // stage 3: exp + partial sums (all threads)
    {
      float mm = mrow[srow];
      float ps = 0.f;
#pragma unroll
      for (int j = 0; j < 8; ++j) {
        float p = __expf(Sbuf[srow][sseg * 8 + j] - mm);
        Pbuf[srow][sseg * 8 + j] = (_Float16)p;
        ps += p;
      }
      red[srow][sseg] = ps;
    }
    __syncthreads();
    // stage 4: finalize running denominator (16 threads)
    if (threadIdx.x < 16) {
      int row = threadIdx.x;
      float sum = 0.f;
#pragma unroll
      for (int j = 0; j < 16; ++j) sum += red[row][j];
      lrow[row] = lrow[row] * arow[row] + sum;
    }
    __syncthreads();

    // ---- rescale accumulators, then O += P @ V over this chunk ----
    float al[8];
#pragma unroll
    for (int r = 0; r < 8; ++r) al[r] = arow[r + half * 8];
#pragma unroll
    for (int nt = 0; nt < 4; ++nt)
#pragma unroll
      for (int r = 0; r < 8; ++r) o[nt][r] *= al[r];

#pragma unroll
    for (int kt = 0; kt < 4; ++kt) {          // 32 keys per WMMA K-step
      v16h pa;
      const v8h* pp = (const v8h*)(&Pbuf[l16][kt * 32 + half * 16]);
      v8h plo = pp[0];
      v8h phi = pp[1];
#pragma unroll
      for (int i = 0; i < 8; ++i) { pa[i] = plo[i]; pa[i + 8] = phi[i]; }
      int kk = c0 + kt * 32 + half * 16;
#pragma unroll
      for (int nt = 0; nt < 4; ++nt) {
        const v16h* vp = (const v16h*)(vbase + (size_t)(d0 + nt * 16 + l16) * SEQ + kk);
        v16h vb = vp[0];
        o[nt] = __builtin_amdgcn_wmma_f32_16x16x32_f16(false, pa, false, vb,
                                                       (short)0, o[nt], false, false);
      }
    }
    __syncthreads();                          // protect Pbuf/arow for next chunk
  }

  // ---- normalize by running denominator and store fp32 output ----
  float li[8];
#pragma unroll
  for (int r = 0; r < 8; ++r) li[r] = 1.f / lrow[r + half * 8];
#pragma unroll
  for (int nt = 0; nt < 4; ++nt) {
#pragma unroll
    for (int r = 0; r < 8; ++r) {
      int m  = q0 + r + half * 8;
      int dc = d0 + nt * 16 + l16;
      O[((size_t)b * SEQ + m) * DIM + dc] = o[nt][r] * li[r];
    }
  }
}

// ---------------------------------------------------------------------------
extern "C" void kernel_launch(void* const* d_in, const int* in_sizes, int n_in,
                              void* d_out, int out_size, void* d_ws, size_t ws_size,
                              hipStream_t stream)
{
  const float* q  = (const float*)d_in[0];
  const float* k  = (const float*)d_in[1];
  const float* v  = (const float*)d_in[2];
  const float* wq = (const float*)d_in[3];
  const float* wk = (const float*)d_in[4];
  float* out = (float*)d_out;

  const size_t NTOK = (size_t)BATCH * SEQ * DIM;   // 8.4M elements
  _Float16* qh = (_Float16*)d_ws;                  // 16 MB
  _Float16* kh = qh + NTOK;                        // 16 MB
  _Float16* vT = kh + NTOK;                        // 16 MB

  // projections: 512x8 macro-tiles (32x64) -> 4096 waves -> 512 blocks, y={Q,K}
  dpa_proj_kernel<<<dim3(512, 2), 256, 0, stream>>>(q, k, wq, wk, qh, kh);
  // V -> f16 transposed copy
  dpa_vconv_kernel<<<dim3(SEQ / 32, DIM / 32, BATCH), 256, 0, stream>>>(v, vT);
  // flash attention: one block per (16 queries, batch)
  dpa_flash_kernel<<<dim3(SEQ / 16, BATCH), 256, 0, stream>>>(qh, kh, vT, out);
}
